// SelfAttention_12300786336257
// MI455X (gfx1250) — matrix operand order repaired and checked by reading, not compiled, not run on hardware
//
#include <hip/hip_runtime.h>
#include <hip/hip_bf16.h>

// Performer/FAVOR+ attention for MI455X (gfx1250), wave32 + WMMA bf16.
// All LDS tiles staged as bf16 in fragment-native major order:
//   A tiles [m][k] (row-major), B tiles [n][k] (transposed) -> every WMMA
//   operand is two contiguous ds_load_b128 per lane. LDS rows padded to
//   40/72 halfwords (16B-aligned rows, non-pow2 bank rotation).
// Workspace (~296 MB): value/qpos/kpos/av bf16 (64 MB each), norms fp32,
// kvs fp32, kvs split-K partials fp32.

namespace {
constexpr int kB  = 4;
constexpr int kL  = 8192;
constexpr int kD  = 1024;
constexpr int kH  = 16;
constexpr int kS  = 64;
constexpr int kM  = 128;
constexpr int kM2 = 256;
constexpr int kHS = kH * kS;            // 1024
constexpr int kRows  = kB * kL;         // 32768
constexpr int kRowsH = kRows * kH;      // 524288
constexpr int kLChunks = 8;
constexpr int kLPerChunk = kL / kLChunks;  // 1024
constexpr int kLd32 = 40;  // padded LDS row (K=32 tiles), 80B rows
constexpr int kLd64 = 72;  // padded LDS row (K=64 tiles), 144B rows
}  // namespace

typedef __bf16 bf16_t;
typedef __attribute__((ext_vector_type(16))) __bf16 v16bf;
typedef __attribute__((ext_vector_type(8)))  __bf16 v8bf;
typedef __attribute__((ext_vector_type(8)))  float  v8f;

__device__ __forceinline__ v8f wmma_bf16(v16bf a, v16bf b, v8f c) {
  // D = A(16x32 bf16) * B(32x16 bf16) + C(16x16 f32)
  return __builtin_amdgcn_wmma_f32_16x16x32_bf16(false, a, false, b,
                                                 (short)0, c, false, false);
}

// A fragment from row-major bf16 LDS tile sm[m][k] (ld halfwords).
// Lane g=lane>>4,m=lane&15; elem j -> k = g*8 + j (j<8), +16 for j>=8.
// => two contiguous 16B runs => 2x ds_load_b128.
__device__ __forceinline__ v16bf frag_a_bf(const bf16_t* sm, int ld, int row0,
                                           int k0) {
  const int lane = threadIdx.x & 31;
  const int g = lane >> 4, m = lane & 15;
  const bf16_t* base = sm + (size_t)(row0 + m) * ld + k0 + g * 8;
  const v8bf lo = *reinterpret_cast<const v8bf*>(base);
  const v8bf hi = *reinterpret_cast<const v8bf*>(base + 16);
  return __builtin_shufflevector(lo, hi, 0, 1, 2, 3, 4, 5, 6, 7, 8, 9, 10, 11,
                                 12, 13, 14, 15);
}

// B fragment from TRANSPOSED bf16 LDS tile smT[n][k] (ld halfwords).
// Lane g=lane>>4,n=lane&15; elem j -> k = g*8 + j (j<8), +16 for j>=8.
__device__ __forceinline__ v16bf frag_b_bf(const bf16_t* smT, int ld, int col0,
                                           int k0) {
  const int lane = threadIdx.x & 31;
  const int g = lane >> 4, n = lane & 15;
  const bf16_t* base = smT + (size_t)(col0 + n) * ld + k0 + g * 8;
  const v8bf lo = *reinterpret_cast<const v8bf*>(base);
  const v8bf hi = *reinterpret_cast<const v8bf*>(base + 16);
  return __builtin_shufflevector(lo, hi, 0, 1, 2, 3, 4, 5, 6, 7, 8, 9, 10, 11,
                                 12, 13, 14, 15);
}

// ---- k1: fused value/pos/slope projections + per-head transforms + norms ----
// grid (rows/64, H), 128 thr (4 waves), each wave a 16-row stripe x 64 cols
__global__ __launch_bounds__(128) void k1_proj_fused(
    const float* __restrict__ src, const float* __restrict__ pos,
    const float* __restrict__ slope, const float* __restrict__ wval,
    const float* __restrict__ wpos, const float* __restrict__ scale,
    const float* __restrict__ offs, bf16_t* __restrict__ value_o,
    bf16_t* __restrict__ qpos_o, bf16_t* __restrict__ kpos_o,
    float* __restrict__ norms_o) {
  __shared__ bf16_t sA[3][64 * kLd32];   // A tiles [row][k]
  __shared__ bf16_t sW[2][64 * kLd32];   // B tiles transposed [n][k]
  const int tid = threadIdx.x, lane = tid & 31, wid = tid >> 5;
  const int row0 = blockIdx.x * 64;
  const int h = blockIdx.y;
  const int stripe = wid * 16;

  v8f accV[4] = {}, accP[4] = {}, accS[4] = {};

  for (int kt = 0; kt < kD / 32; ++kt) {
    const int k0 = kt * 32;
    __syncthreads();
    // A tiles: 64x32 f32 (float4 reads, bf16 row-major stores)
    for (int i = tid; i < 512; i += 128) {
      const int r = i >> 3, c = i & 7;
      const size_t g = (size_t)(row0 + r) * kD + k0 + c * 4;
      const float4 a0 = *reinterpret_cast<const float4*>(src + g);
      const float4 a1 = *reinterpret_cast<const float4*>(pos + g);
      const float4 a2 = *reinterpret_cast<const float4*>(slope + g);
      bf16_t* d0 = &sA[0][(size_t)r * kLd32 + c * 4];
      bf16_t* d1 = &sA[1][(size_t)r * kLd32 + c * 4];
      bf16_t* d2 = &sA[2][(size_t)r * kLd32 + c * 4];
      d0[0] = (bf16_t)a0.x; d0[1] = (bf16_t)a0.y;
      d0[2] = (bf16_t)a0.z; d0[3] = (bf16_t)a0.w;
      d1[0] = (bf16_t)a1.x; d1[1] = (bf16_t)a1.y;
      d1[2] = (bf16_t)a1.z; d1[3] = (bf16_t)a1.w;
      d2[0] = (bf16_t)a2.x; d2[1] = (bf16_t)a2.y;
      d2[2] = (bf16_t)a2.z; d2[3] = (bf16_t)a2.w;
    }
    // B tiles: global [k][n] -> LDS transposed [n][k]
    for (int i = tid; i < 512; i += 128) {
      const int r = i >> 4, c = i & 15;  // r = k row, c*4 = n base
      const size_t g = (size_t)(k0 + r) * kHS + h * kS + c * 4;
      const float4 w0 = *reinterpret_cast<const float4*>(wval + g);
      const float4 w1 = *reinterpret_cast<const float4*>(wpos + g);
      sW[0][(size_t)(c * 4 + 0) * kLd32 + r] = (bf16_t)w0.x;
      sW[0][(size_t)(c * 4 + 1) * kLd32 + r] = (bf16_t)w0.y;
      sW[0][(size_t)(c * 4 + 2) * kLd32 + r] = (bf16_t)w0.z;
      sW[0][(size_t)(c * 4 + 3) * kLd32 + r] = (bf16_t)w0.w;
      sW[1][(size_t)(c * 4 + 0) * kLd32 + r] = (bf16_t)w1.x;
      sW[1][(size_t)(c * 4 + 1) * kLd32 + r] = (bf16_t)w1.y;
      sW[1][(size_t)(c * 4 + 2) * kLd32 + r] = (bf16_t)w1.z;
      sW[1][(size_t)(c * 4 + 3) * kLd32 + r] = (bf16_t)w1.w;
    }
    __syncthreads();
    const v16bf aS = frag_a_bf(sA[0], kLd32, stripe, 0);
    const v16bf aP = frag_a_bf(sA[1], kLd32, stripe, 0);
    const v16bf aX = frag_a_bf(sA[2], kLd32, stripe, 0);
#pragma unroll
    for (int t = 0; t < 4; ++t) {
      const v16bf bV = frag_b_bf(sW[0], kLd32, t * 16, 0);
      const v16bf bP = frag_b_bf(sW[1], kLd32, t * 16, 0);
      accV[t] = wmma_bf16(aS, bV, accV[t]);
      accP[t] = wmma_bf16(aP, bP, accP[t]);
      accS[t] = wmma_bf16(aX, bP, accS[t]);
    }
  }

  const float sc = scale[h];
  const float of = offs[h];
  const int g = lane >> 4, n = lane & 15;
  float sq[8] = {};
#pragma unroll
  for (int t = 0; t < 4; ++t) {
#pragma unroll
    for (int r = 0; r < 8; ++r) {
      const int m = g * 8 + r;
      const size_t row = (size_t)(row0 + stripe + m);
      const size_t idx = row * kHS + h * kS + t * 16 + n;
      const float v = accV[t][r], p = accP[t][r], s = accS[t][r];
      value_o[idx] = (bf16_t)v;
      const float q = sc * p;                   // query_pos
      qpos_o[idx] = (bf16_t)q;
      kpos_o[idx] = (bf16_t)(q + of * sc * s);  // key_pos
      sq[r] += s * s;                           // ||slope_proj|| (pre-scale)
    }
  }
#pragma unroll
  for (int r = 0; r < 8; ++r) {
    float v = sq[r];
    v += __shfl_xor(v, 1, 32);
    v += __shfl_xor(v, 2, 32);
    v += __shfl_xor(v, 4, 32);
    v += __shfl_xor(v, 8, 32);  // reduce across 16 lanes of this half-wave
    if (n == 0) {
      const int m = g * 8 + r;
      norms_o[(size_t)(row0 + stripe + m) * kH + h] = sqrtf(v) / (float)kL;
    }
  }
}

// ---- k2: FAVOR fourier features: dash GEMM [64x64]x[64x128] + sin/cos ----
__global__ __launch_bounds__(128) void k2_fourier(
    const bf16_t* __restrict__ xpos,   // [rowsH][64]
    const float* __restrict__ proj,    // [128][64]  (already [n][k])
    float* __restrict__ prime) {       // [rowsH][256] in d_out
  __shared__ bf16_t sX[64 * kLd64];
  __shared__ bf16_t sP[128 * kLd64];
  const int tid = threadIdx.x, lane = tid & 31, wid = tid >> 5;
  const int row0 = blockIdx.x * 64, stripe = wid * 16;

  {
    const unsigned int* gx =
        reinterpret_cast<const unsigned int*>(xpos + (size_t)row0 * kS);
    unsigned int* sx = reinterpret_cast<unsigned int*>(sX);
    for (int i = tid; i < 64 * 32; i += 128) {  // 32 dwords per 64-elem row
      const int r = i >> 5, c = i & 31;
      sx[(size_t)r * (kLd64 / 2) + c] = gx[(size_t)r * 32 + c];
    }
    for (int i = tid; i < 128 * 64; i += 128) {
      const int r = i >> 6, c = i & 63;
      sP[(size_t)r * kLd64 + c] = (bf16_t)proj[i];
    }
  }
  __syncthreads();

  v8f acc[8] = {};
#pragma unroll
  for (int kt = 0; kt < 2; ++kt) {
    const v16bf aX = frag_a_bf(sX, kLd64, stripe, kt * 32);
#pragma unroll
    for (int t = 0; t < 8; ++t) {
      const v16bf bP = frag_b_bf(sP, kLd64, t * 16, kt * 32);
      acc[t] = wmma_bf16(aX, bP, acc[t]);
    }
  }

  const float cnorm = 0.3535533905932738f;   // 64^-0.25 (data_normalizer)
  const float ratio = 0.08838834764831845f;  // 128^-0.5
  const int g = lane >> 4, n = lane & 15;
#pragma unroll
  for (int t = 0; t < 8; ++t) {
#pragma unroll
    for (int r = 0; r < 8; ++r) {
      const int m = g * 8 + r;
      const size_t row = (size_t)(row0 + stripe + m);
      const float dash = cnorm * acc[t][r];
      float sv, cv;
      __sincosf(dash, &sv, &cv);
      prime[row * kM2 + t * 16 + n]      = ratio * sv;
      prime[row * kM2 + kM + t * 16 + n] = ratio * cv;
    }
  }
}

// ---- k3: kvs partial = K'^T V over one L-chunk; grid (B*H, chunks) ----
__global__ __launch_bounds__(512) void k3_kvs_partial(
    const float* __restrict__ kprime, const bf16_t* __restrict__ value,
    float* __restrict__ part) {
  __shared__ bf16_t sK[256 * kLd32];  // A tile [m][l] (transpose-staged)
  __shared__ bf16_t sV[64 * kLd32];   // B tile [s][l] (transpose-staged)
  const int tid = threadIdx.x, lane = tid & 31, wid = tid >> 5;
  const int bh = blockIdx.x, chunk = blockIdx.y;
  const int b = bh >> 4, h = bh & 15;
  const int mbase = wid * 16;  // 16 waves cover 256 feature rows

  v8f acc[4] = {};
  const int lbeg = chunk * kLPerChunk;
  for (int lt = 0; lt < kLPerChunk / 32; ++lt) {
    const int l0 = lbeg + lt * 32;
    __syncthreads();
    for (int i = tid; i < 32 * 256; i += 512) {
      const int r = i >> 8, c = i & 255;  // r = l, c = m
      sK[(size_t)c * kLd32 + r] =
          (bf16_t)kprime[((size_t)(b * kL + l0 + r) * kH + h) * kM2 + c];
    }
    for (int i = tid; i < 32 * 64; i += 512) {
      const int r = i >> 6, c = i & 63;  // r = l, c = s
      sV[(size_t)c * kLd32 + r] =
          value[((size_t)(b * kL + l0 + r) * kH + h) * kS + c];
    }
    if (lt + 1 < kLPerChunk / 32) {  // gfx1250 global_prefetch for next tile
      __builtin_prefetch(&kprime[((size_t)(b * kL + l0 + 32) * kH + h) * kM2],
                         0, 0);
    }
    __syncthreads();
    const v16bf aK = frag_a_bf(sK, kLd32, mbase, 0);
#pragma unroll
    for (int t = 0; t < 4; ++t) {
      const v16bf bV = frag_b_bf(sV, kLd32, t * 16, 0);
      acc[t] = wmma_bf16(aK, bV, acc[t]);
    }
  }

  const int g = lane >> 4, n = lane & 15;
  float* dst = part + ((size_t)chunk * (kB * kH) + bh) * (kM2 * kS);
#pragma unroll
  for (int t = 0; t < 4; ++t) {
#pragma unroll
    for (int r = 0; r < 8; ++r) {
      const int m = mbase + g * 8 + r;
      dst[(size_t)m * kS + t * 16 + n] = acc[t][r];
    }
  }
}

// ---- k4: deterministic split-K reduce of kvs partials ----
__global__ __launch_bounds__(256) void k4_kvs_reduce(
    const float* __restrict__ part, float* __restrict__ kvs) {
  const size_t n = (size_t)kB * kH * kM2 * kS;
  const size_t i = (size_t)blockIdx.x * blockDim.x + threadIdx.x;
  if (i >= n) return;
  float s = 0.f;
#pragma unroll
  for (int c = 0; c < kLChunks; ++c) s += part[(size_t)c * n + i];
  kvs[i] = s;
}

// ---- k5: av = Q' * kvs, scaled by norms; grid (B*H, L/64) ----
__global__ __launch_bounds__(128) void k5_av(
    const float* __restrict__ qprime, const float* __restrict__ kvs,
    const float* __restrict__ norms, bf16_t* __restrict__ av) {
  __shared__ bf16_t sQ[64 * kLd32];   // A tile [l][k]
  __shared__ bf16_t sKV[64 * kLd32];  // B tile [s][k] (transpose-staged)
  const int tid = threadIdx.x, lane = tid & 31, wid = tid >> 5;
  const int bh = blockIdx.x;
  const int b = bh >> 4, h = bh & 15;
  const int l0 = blockIdx.y * 64, stripe = wid * 16;

  v8f acc[4] = {};
  for (int kt = 0; kt < kM2 / 32; ++kt) {
    const int k0 = kt * 32;
    __syncthreads();
    for (int i = tid; i < 64 * 32; i += 128) {
      const int r = i >> 5, c = i & 31;  // r = l, c = k
      sQ[(size_t)r * kLd32 + c] =
          (bf16_t)qprime[((size_t)(b * kL + l0 + r) * kH + h) * kM2 + k0 + c];
    }
    for (int i = tid; i < 32 * 64; i += 128) {
      const int r = i >> 6, c = i & 63;  // r = k, c = s
      sKV[(size_t)c * kLd32 + r] =
          (bf16_t)kvs[(size_t)bh * (kM2 * kS) + (size_t)(k0 + r) * kS + c];
    }
    __syncthreads();
    const v16bf aQ = frag_a_bf(sQ, kLd32, stripe, 0);
#pragma unroll
    for (int t = 0; t < 4; ++t) {
      const v16bf bK = frag_b_bf(sKV, kLd32, t * 16, 0);
      acc[t] = wmma_bf16(aQ, bK, acc[t]);
    }
  }

  const int g = lane >> 4, n = lane & 15;
#pragma unroll
  for (int r = 0; r < 8; ++r) {
    const int m = g * 8 + r;
    const int l = l0 + stripe + m;
    const float nrm = norms[(size_t)(b * kL + l) * kH + h];
#pragma unroll
    for (int t = 0; t < 4; ++t) {
      av[(size_t)(b * kL + l) * kHS + h * kS + t * 16 + n] =
          (bf16_t)(nrm * acc[t][r]);
    }
  }
}

// ---- k6: out = av @ output_weight^T; grid (rows/64, D/64) ----
__global__ __launch_bounds__(128) void k6_out(
    const bf16_t* __restrict__ av, const float* __restrict__ wout,
    float* __restrict__ out) {
  __shared__ bf16_t sA[64 * kLd32];  // A tile [row][k]
  __shared__ bf16_t sW[64 * kLd32];  // wout [d][k] is already [n][k]
  const int tid = threadIdx.x, lane = tid & 31, wid = tid >> 5;
  const int row0 = blockIdx.x * 64, d0 = blockIdx.y * 64;
  const int stripe = wid * 16;

  v8f acc[4] = {};
  for (int kt = 0; kt < kHS / 32; ++kt) {
    const int k0 = kt * 32;
    __syncthreads();
    {
      unsigned int* sa = reinterpret_cast<unsigned int*>(sA);
      const unsigned int* ga = reinterpret_cast<const unsigned int*>(av);
      for (int i = tid; i < 64 * 16; i += 128) {  // 16 dwords per 32-elem row
        const int r = i >> 4, c = i & 15;
        sa[(size_t)r * (kLd32 / 2) + c] =
            ga[(size_t)(row0 + r) * (kHS / 2) + (k0 >> 1) + c];
      }
    }
    for (int i = tid; i < 64 * 32; i += 128) {
      const int r = i >> 5, c = i & 31;  // r = d, c = k
      sW[(size_t)r * kLd32 + c] = (bf16_t)wout[(size_t)(d0 + r) * kHS + k0 + c];
    }
    __syncthreads();
    const v16bf aA = frag_a_bf(sA, kLd32, stripe, 0);
#pragma unroll
    for (int t = 0; t < 4; ++t) {
      const v16bf bW = frag_b_bf(sW, kLd32, t * 16, 0);
      acc[t] = wmma_bf16(aA, bW, acc[t]);
    }
  }

  const int g = lane >> 4, n = lane & 15;
#pragma unroll
  for (int t = 0; t < 4; ++t) {
#pragma unroll
    for (int r = 0; r < 8; ++r) {
      const int m = g * 8 + r;
      out[(size_t)(row0 + stripe + m) * kD + d0 + t * 16 + n] = acc[t][r];
    }
  }
}

extern "C" void kernel_launch(void* const* d_in, const int* in_sizes, int n_in,
                              void* d_out, int out_size, void* d_ws,
                              size_t ws_size, hipStream_t stream) {
  (void)in_sizes; (void)n_in; (void)out_size; (void)ws_size;
  const float* src   = (const float*)d_in[0];  // source_input [B,L,D]
  const float* pos   = (const float*)d_in[1];  // pos_ft [B,L,D]
  const float* slope = (const float*)d_in[2];  // pos_ft_slopes [B,L,D]
  const float* wval  = (const float*)d_in[3];  // value_weight [D,H,S]
  const float* wpos  = (const float*)d_in[4];  // pos_ft_weight [D,H,S]
  const float* scale = (const float*)d_in[5];  // pos_ft_scale [H]
  const float* offs  = (const float*)d_in[6];  // pos_ft_offsets [H]
  const float* wout  = (const float*)d_in[7];  // output_weight [D,H*S]
  const float* proj  = (const float*)d_in[8];  // projection_matrix [M,S]

  float* out    = (float*)d_out;                       // [B,L,D]
  float* qprime = out + (size_t)kRows * kD;            // [B,L,H,2M]
  float* kprime = qprime + (size_t)kRowsH * kM2;       // [B,L,H,2M]

  char* ws = (char*)d_ws;
  size_t off = 0;
  auto wsalloc = [&](size_t bytes) -> char* {
    char* p = ws + off;
    off += (bytes + 255) & ~(size_t)255;
    return p;
  };
  bf16_t* value = (bf16_t*)wsalloc((size_t)kRows * kHS * sizeof(bf16_t));
  bf16_t* qpos  = (bf16_t*)wsalloc((size_t)kRows * kHS * sizeof(bf16_t));
  bf16_t* kpos  = (bf16_t*)wsalloc((size_t)kRows * kHS * sizeof(bf16_t));
  float*  norms = (float*)wsalloc((size_t)kRowsH * sizeof(float));
  float*  kvs   = (float*)wsalloc((size_t)kB * kH * kM2 * kS * sizeof(float));
  float*  part  = (float*)wsalloc(
      (size_t)kLChunks * kB * kH * kM2 * kS * sizeof(float));
  bf16_t* avb   = (bf16_t*)wsalloc((size_t)kRows * kHS * sizeof(bf16_t));

  k1_proj_fused<<<dim3(kRows / 64, kH), dim3(128), 0, stream>>>(
      src, pos, slope, wval, wpos, scale, offs, value, qpos, kpos, norms);
  k2_fourier<<<dim3(kRowsH / 64), dim3(128), 0, stream>>>(qpos, proj, qprime);
  k2_fourier<<<dim3(kRowsH / 64), dim3(128), 0, stream>>>(kpos, proj, kprime);
  k3_kvs_partial<<<dim3(kB * kH, kLChunks), dim3(512), 0, stream>>>(
      kprime, value, part);
  k4_kvs_reduce<<<dim3((kB * kH * kM2 * kS + 255) / 256), dim3(256), 0,
                  stream>>>(part, kvs);
  k5_av<<<dim3(kB * kH, kL / 64), dim3(128), 0, stream>>>(qprime, kvs, norms,
                                                          avb);
  k6_out<<<dim3(kRows / 64, kD / 64), dim3(128), 0, stream>>>(avb, wout, out);
}
